// StandardAttention_19198503813766
// MI455X (gfx1250) — matrix operand order repaired and checked by reading, not compiled, not run on hardware
//
#include <hip/hip_runtime.h>
#include <hip/hip_bf16.h>

// ---------------------------------------------------------------------------
// StandardAttention (B=1, S=4096, H=1024, NH=16, HD=64) for MI455X / gfx1250.
// All GEMM work (QKV proj, Q*K^T, P*V, O proj) on v_wmma_f32_16x16x32_bf16
// (wave32, fp32 accumulate). Tiles staged LDS via the Tensor Data Mover
// (tensor_load_to_lds + s_wait_tensorcnt) with double buffering; softmax and
// RoPE in fp32 VALU.
// ---------------------------------------------------------------------------

typedef __bf16 bf16_t;
typedef __attribute__((ext_vector_type(16))) __bf16 v16bf;
typedef __attribute__((ext_vector_type(8)))  __bf16 v8bf;
typedef __attribute__((ext_vector_type(8)))  float  v8f;
typedef __attribute__((ext_vector_type(4)))  unsigned int v4u;
typedef __attribute__((ext_vector_type(8)))  int v8i;
typedef __attribute__((ext_vector_type(4)))  int v4i;

union frag16 { v16bf v; v8bf h[2]; };

constexpr int S  = 4096;
constexpr int Hd = 1024;
constexpr int NH = 16;
constexpr int HD = 64;

#define HAS_TDM __has_builtin(__builtin_amdgcn_tensor_load_to_lds)

#if HAS_TDM
// ---------------------------------------------------------------------------
// TDM: DMA a 2D tile (rows x cols of bf16) from global into LDS, padding the
// LDS destination every (1<<(pad_interval+1)) DWORDs by (pad_amount+1) DWORDs
// so LDS rows land on the padded strides the WMMA fragment loads expect.
// Wave-level op: issue from one wave, EXEC ignored, tracked by TENSORcnt.
// ---------------------------------------------------------------------------
__device__ __forceinline__ void tdm_load_2d(
    void* lds_dst, const void* gsrc,
    unsigned tensor_d0, unsigned tensor_d1,   // elements / rows (OOB bounds)
    unsigned tile_d0, unsigned tile_d1,       // tile elements / rows
    unsigned stride0,                         // row stride in elements
    unsigned pad_interval, unsigned pad_amount) {
  const unsigned lds_addr = (unsigned)(unsigned long long)lds_dst;
  const unsigned long long ga = (unsigned long long)gsrc;
  const unsigned pad_en = pad_amount ? 1u : 0u;
  v4u g0;
  g0[0] = 1u;                                            // count=1, user mode
  g0[1] = lds_addr;                                      // LDS byte address
  g0[2] = (unsigned)ga;                                  // global_addr[31:0]
  g0[3] = (unsigned)((ga >> 32) & 0x01ffffffu) | (2u << 30);  // type=2 image
  v8i g1;
  g1[0] = (int)((1u << 16) |                             // data_size = 2B
                (pad_en << 20) | (pad_interval << 22) | (pad_amount << 25));
  g1[1] = (int)((tensor_d0 & 0xffffu) << 16);            // abar=0 | dim0.lo
  g1[2] = (int)((tensor_d0 >> 16) | ((tensor_d1 & 0xffffu) << 16));
  g1[3] = (int)((tensor_d1 >> 16) | (tile_d0 << 16));
  g1[4] = (int)tile_d1;                                  // tile_dim2 = 0
  g1[5] = (int)stride0;                                  // dim0_stride[31:0]
  g1[6] = 0;                                             // hi bits, dim1_stride
  g1[7] = 0;
  v4i gz = {};
#if defined(__clang_major__) && (__clang_major__ >= 23)
  v8i gz8 = {};
  __builtin_amdgcn_tensor_load_to_lds(g0, g1, gz, gz, gz8, 0);
#else
  __builtin_amdgcn_tensor_load_to_lds(g0, g1, gz, gz, 0);
#endif
}
#endif  // HAS_TDM

// ---- A-fragment (16x32 bf16): lanes 0-15 hold K {0..7,16..23}, lanes 16-31
// hold K {8..15,24..31} (ISA 7.12.2). rowptr = row base at current k-step.
__device__ __forceinline__ v16bf load_fragA(const bf16_t* rowptr, int half) {
  frag16 f;
  const int klo = half * 8;
  f.h[0] = *reinterpret_cast<const v8bf*>(rowptr + klo);
  f.h[1] = *reinterpret_cast<const v8bf*>(rowptr + 16 + klo);
  return f.v;
}
// ---- B-fragment (32x16 bf16): lane n holds column n; lanes 0-15 K {0..7,16..23},
// lanes 16-31 K {8..15,24..31}.
__device__ __forceinline__ v16bf load_fragB(const bf16_t* rowptr, int half) {
  frag16 f;
  const int kb = half * 8;
  f.h[0] = *reinterpret_cast<const v8bf*>(rowptr + kb);
  f.h[1] = *reinterpret_cast<const v8bf*>(rowptr + 16 + kb);
  return f.v;
}

// ---------------------------------------------------------------------------
// fp32 -> bf16 cast (4 elements / thread)
// ---------------------------------------------------------------------------
__global__ void cast_f32_bf16(const float* __restrict__ src,
                              bf16_t* __restrict__ dst, int n) {
  int i = (blockIdx.x * blockDim.x + threadIdx.x) * 4;
  if (i + 3 < n) {
    float4 f = *reinterpret_cast<const float4*>(src + i);
    dst[i + 0] = (bf16_t)f.x;
    dst[i + 1] = (bf16_t)f.y;
    dst[i + 2] = (bf16_t)f.z;
    dst[i + 3] = (bf16_t)f.w;
  }
}

// ---------------------------------------------------------------------------
// NT GEMM: C[M,N] = A[M,K] * B[N,K]^T  (A,B bf16 row-major, K-contiguous).
// Block tile 128x128, 8 waves, each wave 64x32 (4x2 accs of 16x16).
// K-tiles of 32 staged in double-buffered LDS by the TDM; wave 0 drives the
// DMA pipeline (issue next tile, s_wait_tensorcnt, one barrier per step).
// storeMode 0: out bf16 head-major [head][row][d]; storeMode 1: fp32 [M,N].
// ---------------------------------------------------------------------------
constexpr int LDT = 48; // LDS row stride (bf16): 96B rows, 16B/32B aligned
__global__ __launch_bounds__(256) void gemm_bf16_nt(
    const bf16_t* __restrict__ A, const bf16_t* __restrict__ Bm,
    bf16_t* __restrict__ outHM, float* __restrict__ outF,
    int M, int N, int K, int storeMode) {
  __shared__ bf16_t As[2][128 * LDT];
  __shared__ bf16_t Bs[2][128 * LDT];

  const int tid  = threadIdx.x;
  const int wave = tid >> 5;
  const int lane = tid & 31;
  const int half = lane >> 4;
  const int lrow = lane & 15;
  const int wm = wave >> 2;            // 0..1  -> M offset wm*64
  const int wn = wave & 3;             // 0..3  -> N offset wn*32
  const int bm0 = blockIdx.y * 128;
  const int bn0 = blockIdx.x * 128;

  v8f acc[4][2] = {};

#if HAS_TDM
  if (wave == 0) {  // prime the pipeline: first 128x32 tiles into buffer 0
    tdm_load_2d(&As[0][0], &A[(size_t)bm0 * K],  K, M, 32, 128, K, 3, 7);
    tdm_load_2d(&Bs[0][0], &Bm[(size_t)bn0 * K], K, N, 32, 128, K, 3, 7);
  }
#endif

  int buf = 0;
  for (int k0 = 0; k0 < K; k0 += 32) {
#if HAS_TDM
    if (wave == 0) __builtin_amdgcn_s_wait_tensorcnt(0);
    __syncthreads();  // tile `buf` resident; previous compute drained
    if (wave == 0 && k0 + 32 < K) {   // overlap next DMA with this WMMA burst
      tdm_load_2d(&As[buf ^ 1][0], &A[(size_t)bm0 * K + k0 + 32],
                  K, M, 32, 128, K, 3, 7);
      tdm_load_2d(&Bs[buf ^ 1][0], &Bm[(size_t)bn0 * K + k0 + 32],
                  K, N, 32, 128, K, 3, 7);
    }
#else
    {  // manual staging fallback: 32B/thread per tile via b128 pairs
      const int ldRow = tid >> 1, ldCol = (tid & 1) * 16;
      const uint4* sA = reinterpret_cast<const uint4*>(
          &A[(size_t)(bm0 + ldRow) * K + k0 + ldCol]);
      uint4* dA = reinterpret_cast<uint4*>(&As[buf][ldRow * LDT + ldCol]);
      dA[0] = sA[0]; dA[1] = sA[1];
      const uint4* sB = reinterpret_cast<const uint4*>(
          &Bm[(size_t)(bn0 + ldRow) * K + k0 + ldCol]);
      uint4* dB = reinterpret_cast<uint4*>(&Bs[buf][ldRow * LDT + ldCol]);
      dB[0] = sB[0]; dB[1] = sB[1];
      if (k0 + 32 < K) {
        __builtin_prefetch(&A[(size_t)(bm0 + ldRow) * K + k0 + 32 + ldCol], 0, 0);
        __builtin_prefetch(&Bm[(size_t)(bn0 + ldRow) * K + k0 + 32 + ldCol], 0, 0);
      }
    }
    __syncthreads();
#endif

    const bf16_t* as = As[buf];
    const bf16_t* bs = Bs[buf];
    v16bf bfr[2], afr[4];
#pragma unroll
    for (int ni = 0; ni < 2; ++ni)
      bfr[ni] = load_fragB(&bs[(wn * 32 + ni * 16 + lrow) * LDT], half);
#pragma unroll
    for (int mi = 0; mi < 4; ++mi)
      afr[mi] = load_fragA(&as[(wm * 64 + mi * 16 + lrow) * LDT], half);
#pragma unroll
    for (int mi = 0; mi < 4; ++mi)
#pragma unroll
      for (int ni = 0; ni < 2; ++ni)
        acc[mi][ni] = __builtin_amdgcn_wmma_f32_16x16x32_bf16(
            false, afr[mi], false, bfr[ni], (short)0, acc[mi][ni], false, false);

#if !HAS_TDM
    __syncthreads();
#endif
    buf ^= 1;
  }

  // C layout: VGPR r, lanes 0-15 -> M=r ; lanes 16-31 -> M=r+8 ; N = lane&15.
#pragma unroll
  for (int mi = 0; mi < 4; ++mi) {
#pragma unroll
    for (int ni = 0; ni < 2; ++ni) {
      const int gn = bn0 + wn * 32 + ni * 16 + lrow;
#pragma unroll
      for (int r = 0; r < 8; ++r) {
        const int gm = bm0 + wm * 64 + mi * 16 + r + half * 8;
        const float val = acc[mi][ni][r];
        if (storeMode == 0) {
          const int head = gn >> 6, d = gn & 63;
          outHM[((size_t)head * S + gm) * HD + d] = (bf16_t)val;
        } else {
          outF[(size_t)gm * N + gn] = val;
        }
      }
    }
  }
}

// ---------------------------------------------------------------------------
// In-place RoPE on bf16 Q,K (head-major [NH][S][64]). fp32 math.
// ---------------------------------------------------------------------------
__global__ void rope_kernel(bf16_t* __restrict__ Q, bf16_t* __restrict__ K,
                            const float* __restrict__ cosT,
                            const float* __restrict__ sinT) {
  const size_t idx = (size_t)blockIdx.x * blockDim.x + threadIdx.x; // NH*S*32
  const int d = (int)(idx & 31);
  const size_t hs = idx >> 5;            // head*S + s
  const int s = (int)(hs & (S - 1));
  const size_t base = hs * HD;
  const float c  = cosT[s * HD + d];
  const float sn = sinT[s * HD + d];     // cos/sin duplicated across halves
  const float q1 = (float)Q[base + d], q2 = (float)Q[base + d + 32];
  Q[base + d]      = (bf16_t)(q1 * c - q2 * sn);
  Q[base + d + 32] = (bf16_t)(q2 * c + q1 * sn);
  const float k1 = (float)K[base + d], k2 = (float)K[base + d + 32];
  K[base + d]      = (bf16_t)(k1 * c - k2 * sn);
  K[base + d + 32] = (bf16_t)(k2 * c + k1 * sn);
}

// ---------------------------------------------------------------------------
// Flash attention, causal. grid = (S/64, NH), block = 128 (4 waves).
// Wave w owns query rows [q0+16w, q0+16w+16). Per 64-key tile: K tile DMA'd
// by the TDM (overlapped with the VALU V-transpose staging), scores via WMMA,
// analytic causal mask, online softmax (shfl within 16-lane halves), P
// re-laid out through LDS, PV via WMMA.
// ---------------------------------------------------------------------------
constexpr int LKV = 80;  // LDS row stride (bf16): 160B rows, 16B/32B aligned
__global__ __launch_bounds__(128) void flash_attn(
    const bf16_t* __restrict__ Q, const bf16_t* __restrict__ K,
    const bf16_t* __restrict__ V, bf16_t* __restrict__ O) {
  __shared__ bf16_t Ks[64 * LKV];
  __shared__ bf16_t Vt[64 * LKV];  // transposed: Vt[d][k]
  __shared__ bf16_t Ps[64 * LKV];  // per-wave P strips (rows 16w..16w+15)

  const int head = blockIdx.y;
  const int qt   = blockIdx.x;
  const int q0   = qt * 64;
  const int tid  = threadIdx.x, wave = tid >> 5, lane = tid & 31;
  const int half = lane >> 4, lrow = lane & 15;

  const bf16_t* Qh = Q + (size_t)head * S * HD;
  const bf16_t* Kh = K + (size_t)head * S * HD;
  const bf16_t* Vh = V + (size_t)head * S * HD;

  // Q fragments for this wave's 16 rows, 2 k-steps over HD=64.
  v16bf qf[2];
  {
    const bf16_t* qp = Qh + (size_t)(q0 + wave * 16 + lrow) * HD;
    qf[0] = load_fragA(qp, half);
    qf[1] = load_fragA(qp + 32, half);
  }

  float mrun[8], lrun[8];
  v8f oacc[4] = {};
#pragma unroll
  for (int r = 0; r < 8; ++r) { mrun[r] = -1e30f; lrun[r] = 0.f; }
  const float scale = 0.125f;  // 1/sqrt(HD)

  for (int kt = 0; kt <= qt; ++kt) {
    const int k0 = kt * 64;
#if HAS_TDM
    if (wave == 0)   // 64x64 K tile -> padded LDS rows (pad 8 DW every 32 DW)
      tdm_load_2d(&Ks[0], Kh + (size_t)k0 * HD, HD, S, 64, 64, HD, 4, 7);
#else
    {  // manual K staging: 64B/thread
      const int row = tid >> 1, ch = (tid & 1) * 32;
      const uint4* src = reinterpret_cast<const uint4*>(
          Kh + (size_t)(k0 + row) * HD + ch);
      uint4* dst = reinterpret_cast<uint4*>(&Ks[row * LKV + ch]);
      dst[0] = src[0]; dst[1] = src[1]; dst[2] = src[2]; dst[3] = src[3];
    }
#endif
    for (int e = tid; e < 64 * 64; e += 128) {  // V transposed into LDS
      const int k = e >> 6, d = e & 63;
      Vt[d * LKV + k] = Vh[(size_t)(k0 + k) * HD + d];
    }
#if HAS_TDM
    if (wave == 0) __builtin_amdgcn_s_wait_tensorcnt(0);
#endif
    __syncthreads();

    // ---- scores = Q * K^T ----
    v8f sacc[4] = {};
    v16bf kfr[4][2];
#pragma unroll
    for (int ni = 0; ni < 4; ++ni)
#pragma unroll
      for (int ks = 0; ks < 2; ++ks)
        kfr[ni][ks] = load_fragB(&Ks[(ni * 16 + lrow) * LKV + ks * 32], half);
#pragma unroll
    for (int ni = 0; ni < 4; ++ni)
#pragma unroll
      for (int ks = 0; ks < 2; ++ks)
        sacc[ni] = __builtin_amdgcn_wmma_f32_16x16x32_bf16(
            false, qf[ks], false, kfr[ni][ks], (short)0, sacc[ni], false, false);

    // ---- causal mask + scale + row max (16-lane half reduction) ----
    float rowmax[8];
#pragma unroll
    for (int r = 0; r < 8; ++r) {
      const int grow = q0 + wave * 16 + r + half * 8;
      float mx = -1e30f;
#pragma unroll
      for (int ni = 0; ni < 4; ++ni) {
        const int gcol = k0 + ni * 16 + lrow;
        float v = sacc[ni][r] * scale;
        v = (gcol <= grow) ? v : -1e30f;
        sacc[ni][r] = v;
        mx = fmaxf(mx, v);
      }
#pragma unroll
      for (int off = 1; off < 16; off <<= 1)
        mx = fmaxf(mx, __shfl_xor(mx, off, 32));
      rowmax[r] = mx;
    }

    // ---- online softmax update ----
#pragma unroll
    for (int r = 0; r < 8; ++r) {
      const float mnew  = fmaxf(mrun[r], rowmax[r]);
      const float alpha = __expf(mrun[r] - mnew);
      mrun[r] = mnew;
      float psum = 0.f;
#pragma unroll
      for (int ni = 0; ni < 4; ++ni) {
        const float p = __expf(sacc[ni][r] - mnew);
        sacc[ni][r] = p;
        psum += p;
      }
#pragma unroll
      for (int off = 1; off < 16; off <<= 1)
        psum += __shfl_xor(psum, off, 32);
      lrun[r] = lrun[r] * alpha + psum;
#pragma unroll
      for (int ni = 0; ni < 4; ++ni) oacc[ni][r] *= alpha;
    }

    // ---- P -> LDS (C layout -> A layout round trip, wave-private strip) ----
#pragma unroll
    for (int r = 0; r < 8; ++r)
#pragma unroll
      for (int ni = 0; ni < 4; ++ni)
        Ps[(wave * 16 + r + half * 8) * LKV + ni * 16 + lrow] =
            (bf16_t)sacc[ni][r];

    // ---- O += P * V ----
    v16bf pf[2], vf[2][4];
#pragma unroll
    for (int ks = 0; ks < 2; ++ks) {
      pf[ks] = load_fragA(&Ps[(wave * 16 + lrow) * LKV + ks * 32], half);
#pragma unroll
      for (int ni = 0; ni < 4; ++ni)
        vf[ks][ni] = load_fragB(&Vt[(ni * 16 + lrow) * LKV + ks * 32], half);
    }
#pragma unroll
    for (int ks = 0; ks < 2; ++ks)
#pragma unroll
      for (int ni = 0; ni < 4; ++ni)
        oacc[ni] = __builtin_amdgcn_wmma_f32_16x16x32_bf16(
            false, pf[ks], false, vf[ks][ni], (short)0, oacc[ni], false, false);
    __syncthreads();
  }

  // ---- finalize: O /= l, write bf16 [S][H] with heads re-interleaved ----
#pragma unroll
  for (int ni = 0; ni < 4; ++ni) {
#pragma unroll
    for (int r = 0; r < 8; ++r) {
      const int row = q0 + wave * 16 + r + half * 8;
      const int d   = ni * 16 + lrow;
      O[(size_t)row * Hd + head * HD + d] =
          (bf16_t)(oacc[ni][r] / lrun[r]);
    }
  }
}

// ---------------------------------------------------------------------------
extern "C" void kernel_launch(void* const* d_in, const int* in_sizes, int n_in,
                              void* d_out, int out_size, void* d_ws,
                              size_t ws_size, hipStream_t stream) {
  (void)in_sizes; (void)n_in; (void)out_size; (void)ws_size;
  const float* hs   = (const float*)d_in[0];   // [1,S,H]
  const float* cosT = (const float*)d_in[1];   // [S,HD]
  const float* sinT = (const float*)d_in[2];   // [S,HD]
  /* d_in[3] additive causal mask: applied analytically in flash_attn */
  const float* Wq = (const float*)d_in[4];
  const float* Wk = (const float*)d_in[5];
  const float* Wv = (const float*)d_in[6];
  const float* Wo = (const float*)d_in[7];
  float* out = (float*)d_out;

  char* ws = (char*)d_ws;
  const size_t MB = 1024 * 1024;
  bf16_t* Xb  = (bf16_t*)(ws + 0 * MB);   // [S,H]       8 MiB
  bf16_t* Wqb = (bf16_t*)(ws + 8 * MB);   // [H,H]       2 MiB
  bf16_t* Wkb = (bf16_t*)(ws + 10 * MB);
  bf16_t* Wvb = (bf16_t*)(ws + 12 * MB);
  bf16_t* Wob = (bf16_t*)(ws + 14 * MB);
  bf16_t* Qb  = (bf16_t*)(ws + 16 * MB);  // [NH,S,HD]   8 MiB
  bf16_t* Kb  = (bf16_t*)(ws + 24 * MB);
  bf16_t* Vb  = (bf16_t*)(ws + 32 * MB);
  bf16_t* Ob  = (bf16_t*)(ws + 40 * MB);  // [S,H]       8 MiB

  const int nX = S * Hd, nW = Hd * Hd;
  cast_f32_bf16<<<nX / (4 * 256), 256, 0, stream>>>(hs, Xb, nX);
  cast_f32_bf16<<<nW / (4 * 256), 256, 0, stream>>>(Wq, Wqb, nW);
  cast_f32_bf16<<<nW / (4 * 256), 256, 0, stream>>>(Wk, Wkb, nW);
  cast_f32_bf16<<<nW / (4 * 256), 256, 0, stream>>>(Wv, Wvb, nW);
  cast_f32_bf16<<<nW / (4 * 256), 256, 0, stream>>>(Wo, Wob, nW);

  dim3 ggrid(Hd / 128, S / 128);  // (8, 32)
  gemm_bf16_nt<<<ggrid, 256, 0, stream>>>(Xb, Wqb, Qb, nullptr, S, Hd, Hd, 0);
  gemm_bf16_nt<<<ggrid, 256, 0, stream>>>(Xb, Wkb, Kb, nullptr, S, Hd, Hd, 0);
  gemm_bf16_nt<<<ggrid, 256, 0, stream>>>(Xb, Wvb, Vb, nullptr, S, Hd, Hd, 0);

  rope_kernel<<<(NH * S * 32) / 256, 256, 0, stream>>>(Qb, Kb, cosT, sinT);

  flash_attn<<<dim3(S / 64, NH), 128, 0, stream>>>(Qb, Kb, Vb, Ob);

  gemm_bf16_nt<<<ggrid, 256, 0, stream>>>(Ob, Wob, nullptr, out, S, Hd, Hd, 1);
}
